// SimplexAttention_29678224015490
// MI455X (gfx1250) — matrix operand order repaired and checked by reading, not compiled, not run on hardware
//
#include <hip/hip_runtime.h>
#include <hip/hip_bf16.h>
#include <math.h>

// ---------------------------------------------------------------------------
// SimplexAttention for MI455X (gfx1250, wave32, WMMA)
//   B=2, T=2048, C=1024, H=16, D=64
// Pipeline:
//   1) Q/K/V/K' = f16(x @ W^T)            (WMMA f16 GEMM, fp32->f16 LDS stage)
//   2) flash attention, both branches fused, alpha-combined  -> fp32
//      (K/K' blocks staged via GLOBAL_LOAD_ASYNC_TO_LDS_B128 when available)
//   3) out = attn @ W_O^T                  (same GEMM, fp32 out)
// ---------------------------------------------------------------------------

typedef __attribute__((ext_vector_type(16))) _Float16 v16h;
typedef __attribute__((ext_vector_type(8)))  _Float16 v8h;
typedef __attribute__((ext_vector_type(4)))  _Float16 v4h;
typedef __attribute__((ext_vector_type(8)))  float    v8f;
typedef __attribute__((ext_vector_type(4)))  int      v4i;

#define BATCH_N 2
#define T_SEQ   2048
#define CDIM    1024
#define HEADS_N 16
#define HD      64

#if defined(__AMDGCN__) && __has_builtin(__builtin_amdgcn_global_load_async_to_lds_b128)
#define HAVE_ASYNC_LDS 1
#else
#define HAVE_ASYNC_LDS 0
#endif

#if HAVE_ASYNC_LDS
typedef __attribute__((address_space(1))) v4i* gv4i_p;  // global int4*
typedef __attribute__((address_space(3))) v4i* lv4i_p;  // LDS int4*
static __device__ inline void async_b128(const void* g, void* l) {
  __builtin_amdgcn_global_load_async_to_lds_b128((gv4i_p)g, (lv4i_p)l, 0, 0);
}
static __device__ inline void wait_async0() {
#if __has_builtin(__builtin_amdgcn_s_wait_asynccnt)
  __builtin_amdgcn_s_wait_asynccnt(0);
#else
  asm volatile("s_wait_asynccnt 0x0" ::: "memory");
#endif
}
#endif

static __device__ inline v8f wmma16(v16h a, v16h b, v8f c) {
  // D = A(16x32 f16) * B(32x16 f16) + C(16x16 f32)
  return __builtin_amdgcn_wmma_f32_16x16x32_f16(
      /*neg_a=*/false, a, /*neg_b=*/false, b,
      /*c_mod=*/(short)0, c, /*reuse_a=*/false, /*reuse_b=*/false);
}

// Assemble a v16h fragment from two contiguous 16-byte chunks (ds/global b128).
static __device__ inline v16h ldfrag(const _Float16* p0, const _Float16* p1) {
  v8h a = *(const v8h*)p0;
  v8h b = *(const v8h*)p1;
  v16h r;
#pragma unroll
  for (int i = 0; i < 8; ++i) { r[i] = a[i]; r[i + 8] = b[i]; }
  return r;
}

static __device__ inline float redmax16(float v) {
  v = fmaxf(v, __shfl_xor(v, 1, 32));
  v = fmaxf(v, __shfl_xor(v, 2, 32));
  v = fmaxf(v, __shfl_xor(v, 4, 32));
  v = fmaxf(v, __shfl_xor(v, 8, 32));
  return v;
}
static __device__ inline float redsum16(float v) {
  v += __shfl_xor(v, 1, 32);
  v += __shfl_xor(v, 2, 32);
  v += __shfl_xor(v, 4, 32);
  v += __shfl_xor(v, 8, 32);
  return v;
}

// ---------------------------------------------------------------------------
// GEMM: Y = A(MxK, f32, row-major) @ W(NxK, f32, row-major)^T
// Output: f16 (Yh) or f32 (Yf). Tile 128x128, K-step 32, 8 waves of 32x64.
// ---------------------------------------------------------------------------
__global__ __launch_bounds__(256) void gemm_xwt_kernel(
    const float* __restrict__ A, const float* __restrict__ W,
    _Float16* __restrict__ Yh, float* __restrict__ Yf,
    int M, int N, int K) {
  __shared__ _Float16 sA[128][40];  // stride 80B (multiple of 16B)
  __shared__ _Float16 sB[128][40];

  const int tid  = threadIdx.x;
  const int lane = tid & 31;
  const int wave = tid >> 5;
  const int wm = wave & 3;   // 4 wave-rows  (32 rows each)
  const int wn = wave >> 2;  // 2 wave-cols  (64 cols each)
  const int ln = lane & 15;
  const int hi = (lane >> 4) & 1;
  const int m0 = blockIdx.y * 128;
  const int n0 = blockIdx.x * 128;

  v8f acc[2][4];
#pragma unroll
  for (int i = 0; i < 2; ++i)
#pragma unroll
    for (int j = 0; j < 4; ++j) { v8f z = {}; acc[i][j] = z; }

  const int nk = K >> 5;
  for (int kt = 0; kt < nk; ++kt) {
    const int k0 = kt * 32;
    __syncthreads();
    // Stage fp32 -> f16 tiles (128x32 each): 1024 float4 per tile, 4/thread.
#pragma unroll
    for (int i = 0; i < 4; ++i) {
      const int j   = tid + 256 * i;  // 0..1023
      const int row = j >> 3;
      const int c4  = j & 7;
      const float4 av = *(const float4*)&A[(size_t)(m0 + row) * K + k0 + c4 * 4];
      v4h pa = { (_Float16)av.x, (_Float16)av.y, (_Float16)av.z, (_Float16)av.w };
      *(v4h*)&sA[row][c4 * 4] = pa;
      const float4 wv = *(const float4*)&W[(size_t)(n0 + row) * K + k0 + c4 * 4];
      v4h pb = { (_Float16)wv.x, (_Float16)wv.y, (_Float16)wv.z, (_Float16)wv.w };
      *(v4h*)&sB[row][c4 * 4] = pb;
      if (kt + 1 < nk) {  // WGP-scope prefetch of next k-tile
        __builtin_prefetch(&A[(size_t)(m0 + row) * K + k0 + 32 + c4 * 4], 0, 3);
        __builtin_prefetch(&W[(size_t)(n0 + row) * K + k0 + 32 + c4 * 4], 0, 3);
      }
    }
    __syncthreads();

    // A fragments: 16-bit A 16x32 layout (lanes 0-15: K 0..7 & 16..23; 16-31: +8)
    v16h af[2];
#pragma unroll
    for (int i = 0; i < 2; ++i) {
      const int row  = wm * 32 + i * 16 + ln;
      const int koff = hi * 8;
      af[i] = ldfrag(&sA[row][koff], &sA[row][16 + koff]);
    }
    // B fragments: 16-bit B 32x16 layout (lanes 0-15: K 0..7 & 16..23; 16-31: +8)
#pragma unroll
    for (int j = 0; j < 4; ++j) {
      const int col = wn * 64 + j * 16 + ln;
      const int klo = hi * 8;
      v16h bf = ldfrag(&sB[col][klo], &sB[col][klo + 16]);
#pragma unroll
      for (int i = 0; i < 2; ++i) acc[i][j] = wmma16(af[i], bf, acc[i][j]);
    }
  }

  // C/D layout: VGPR r, lane -> M = r + 8*hi, N = lane%16
#pragma unroll
  for (int i = 0; i < 2; ++i)
#pragma unroll
    for (int j = 0; j < 4; ++j)
#pragma unroll
      for (int r = 0; r < 8; ++r) {
        const int m = m0 + wm * 32 + i * 16 + r + 8 * hi;
        const int n = n0 + wn * 64 + j * 16 + ln;
        const float v = acc[i][j][r];
        if (Yh) Yh[(size_t)m * N + n] = (_Float16)v;
        else    Yf[(size_t)m * N + n] = v;
      }
}

// ---------------------------------------------------------------------------
// Fused dual-branch flash attention. One (b,h,64-query-block) per workgroup,
// 4 waves x 16 query rows. Streams 64-key blocks of K, K', V through LDS.
// ---------------------------------------------------------------------------
__global__ __launch_bounds__(128) void attn_kernel(
    const _Float16* __restrict__ Q,  const _Float16* __restrict__ Ks,
    const _Float16* __restrict__ Kp, const _Float16* __restrict__ V,
    float* __restrict__ Out, const float* __restrict__ alphaPtr) {
  __shared__ _Float16 sK [64][72];      // (key, d)  for scores B-frag
  __shared__ _Float16 sKp[64][72];
  __shared__ _Float16 sVt[64][72];      // (d, key)  for P@V B-frag
  __shared__ _Float16 sP [4][16][72];   // per-wave P staging (A-frag re-entry)

  const int tid  = threadIdx.x;
  const int lane = tid & 31;
  const int wave = tid >> 5;
  const int ln = lane & 15;
  const int hi = (lane >> 4) & 1;
  const int b = blockIdx.y / HEADS_N;
  const int h = blockIdx.y % HEADS_N;
  const size_t baseBT = (size_t)b * T_SEQ;
  const int hcol = h * HD;
  const int q0 = blockIdx.x * 64 + wave * 16;

  // Q fragments for the wave's 16 rows, pre-scaled by 1/sqrt(D) = 0.125
  v16h qf[2];
#pragma unroll
  for (int c = 0; c < 2; ++c) {
    const _Float16* qp = &Q[(baseBT + q0 + ln) * CDIM + hcol];
    const int koff = c * 32 + hi * 8;
    v16h f = ldfrag(qp + koff, qp + koff + 16);
    qf[c] = f * (_Float16)0.125f;
  }

  v8f accO[2][4];
  float mst[2][8], lst[2][8];
#pragma unroll
  for (int br = 0; br < 2; ++br) {
#pragma unroll
    for (int j = 0; j < 4; ++j) { v8f z = {}; accO[br][j] = z; }
#pragma unroll
    for (int r = 0; r < 8; ++r) { mst[br][r] = -1e30f; lst[br][r] = 0.f; }
  }

  for (int kb = 0; kb < T_SEQ / 64; ++kb) {
    __syncthreads();
    // Cooperative load of 64-key blocks. K/K': async direct-to-LDS b128 copies
    // (ASYNCcnt) when the toolchain exposes them; V: staged + transposed.
#pragma unroll
    for (int i = 0; i < 4; ++i) {
      const int j   = tid + 128 * i;  // 0..511
      const int key = j >> 3;
      const int c8  = j & 7;
      const size_t g = (baseBT + kb * 64 + key) * CDIM + hcol + c8 * 8;
#if HAVE_ASYNC_LDS
      async_b128(&Ks[g], &sK [key][c8 * 8]);
      async_b128(&Kp[g], &sKp[key][c8 * 8]);
#else
      *(v8h*)&sK [key][c8 * 8] = *(const v8h*)&Ks[g];
      *(v8h*)&sKp[key][c8 * 8] = *(const v8h*)&Kp[g];
#endif
      const v8h vv = *(const v8h*)&V[g];
#pragma unroll
      for (int e = 0; e < 8; ++e) sVt[c8 * 8 + e][key] = vv[e];
    }
#if HAVE_ASYNC_LDS
    wait_async0();
#endif
    __syncthreads();

    for (int br = 0; br < 2; ++br) {
      const _Float16 (*sKb)[72] = br ? sKp : sK;

      // S = (Q/sqrt(D)) @ K^T : 4 n-tiles x 2 k-chunks
      v8f s[4];
#pragma unroll
      for (int j = 0; j < 4; ++j) {
        v8f z = {};
        const int key = j * 16 + ln;
#pragma unroll
        for (int c = 0; c < 2; ++c) {
          const int dlo = c * 32 + hi * 8;
          v16h bf = ldfrag(&sKb[key][dlo], &sKb[key][dlo + 16]);
          z = wmma16(qf[c], bf, z);
        }
        s[j] = z;
      }

      // Online softmax per row (VGPR r <-> row r + 8*hi)
#pragma unroll
      for (int r = 0; r < 8; ++r) {
        float rowmax = fmaxf(fmaxf(s[0][r], s[1][r]), fmaxf(s[2][r], s[3][r]));
        rowmax = redmax16(rowmax);
        const float mold = mst[br][r];
        const float mnew = fmaxf(mold, rowmax);
        const float corr = __expf(mold - mnew);
        const int prow = r + 8 * hi;
        float rsum = 0.f;
#pragma unroll
        for (int j = 0; j < 4; ++j) {
          const float p = __expf(s[j][r] - mnew);
          rsum += p;
          sP[wave][prow][j * 16 + ln] = (_Float16)p;
          accO[br][j][r] *= corr;
        }
        lst[br][r] = lst[br][r] * corr + redsum16(rsum);
        mst[br][r] = mnew;
      }

      // O += P @ V  (P re-read from per-wave LDS in A-frag layout)
#pragma unroll
      for (int c = 0; c < 2; ++c) {
        const int koff = c * 32 + hi * 8;
        v16h pf = ldfrag(&sP[wave][ln][koff], &sP[wave][ln][koff + 16]);
#pragma unroll
        for (int j = 0; j < 4; ++j) {
          const int d   = j * 16 + ln;
          const int klo = c * 32 + hi * 8;
          v16h vf = ldfrag(&sVt[d][klo], &sVt[d][klo + 16]);
          accO[br][j] = wmma16(pf, vf, accO[br][j]);
        }
      }
    }
  }

  // alpha-combine the two normalized branches, write fp32 (B,T,C)
  const float alpha = *alphaPtr;
#pragma unroll
  for (int j = 0; j < 4; ++j)
#pragma unroll
    for (int r = 0; r < 8; ++r) {
      const int qrow = q0 + r + 8 * hi;
      const int d = j * 16 + ln;
      const float o = alpha * accO[0][j][r] / lst[0][r] +
                      (1.f - alpha) * accO[1][j][r] / lst[1][r];
      Out[(baseBT + qrow) * CDIM + hcol + d] = o;
    }
}

// ---------------------------------------------------------------------------
extern "C" void kernel_launch(void* const* d_in, const int* in_sizes, int n_in,
                              void* d_out, int out_size, void* d_ws, size_t ws_size,
                              hipStream_t stream) {
  const float* x     = (const float*)d_in[0];
  const float* WQ    = (const float*)d_in[1];
  const float* WK    = (const float*)d_in[2];
  const float* WV    = (const float*)d_in[3];
  const float* WKp   = (const float*)d_in[4];
  const float* WO    = (const float*)d_in[5];
  const float* alpha = (const float*)d_in[6];

  const int M = BATCH_N * T_SEQ;  // 4096
  const int C = CDIM;             // 1024

  // Workspace: Q,K,V,K' f16 (4 x 8MB) + attn fp32 (16MB) = 48MB
  _Float16* Qh   = (_Float16*)d_ws;
  _Float16* Kh   = Qh  + (size_t)M * C;
  _Float16* Vh   = Kh  + (size_t)M * C;
  _Float16* Kph  = Vh  + (size_t)M * C;
  float*    attn = (float*)(Kph + (size_t)M * C);

  dim3 gGemm(C / 128, M / 128);  // 8 x 32
  gemm_xwt_kernel<<<gGemm, 256, 0, stream>>>(x, WQ,  Qh,  nullptr, M, C, C);
  gemm_xwt_kernel<<<gGemm, 256, 0, stream>>>(x, WK,  Kh,  nullptr, M, C, C);
  gemm_xwt_kernel<<<gGemm, 256, 0, stream>>>(x, WV,  Vh,  nullptr, M, C, C);
  gemm_xwt_kernel<<<gGemm, 256, 0, stream>>>(x, WKp, Kph, nullptr, M, C, C);

  dim3 gAttn(T_SEQ / 64, BATCH_N * HEADS_N);  // 32 x 32
  attn_kernel<<<gAttn, 128, 0, stream>>>(Qh, Kh, Kph, Vh, attn, alpha);

  gemm_xwt_kernel<<<gGemm, 256, 0, stream>>>(attn, WO, nullptr, (float*)d_out, M, C, C);
}
